// TokenI2MoE_12429635355021
// MI455X (gfx1250) — matrix operand order repaired and checked by reading, not compiled, not run on hardware
//
#include <hip/hip_runtime.h>
#include <hip/hip_bf16.h>

#define BB 16
#define NN 4096
#define CC 256
#define KK 8
#define EPSF 1e-6f
#define RATIO 0.1f
#define TEMPF 1.0f

typedef __attribute__((ext_vector_type(16))) __bf16 v16bf;
typedef __attribute__((ext_vector_type(8)))  __bf16 v8bf;
typedef __attribute__((ext_vector_type(8)))  float  v8f;

// ---------------- workspace layout (bytes) ----------------
constexpr size_t align_up(size_t x) { return (x + 255) & ~size_t(255); }
constexpr size_t OFF_W1BF = 0;                                     // K*C*C bf16
constexpr size_t SZ_W1BF  = (size_t)KK * CC * CC * 2;
constexpr size_t OFF_ZERO = align_up(OFF_W1BF + SZ_W1BF);          // zeroed region:
constexpr size_t OFF_SBUF = OFF_ZERO;                              //  B*K*C f32
constexpr size_t OFF_MASS = OFF_SBUF + (size_t)BB * KK * CC * 4;   //  B*K f32
constexpr size_t OFF_CNT  = OFF_MASS + (size_t)BB * KK * 4;        //  B*K i32
constexpr size_t OFF_USE  = OFF_CNT  + (size_t)BB * KK * 4;        //  K i32
constexpr size_t OFF_CONF = OFF_USE  + (size_t)KK * 4;             //  1 f32
constexpr size_t ZERO_FLOATS = (OFF_CONF + 4 - OFF_ZERO) / 4;
constexpr size_t OFF_GG   = align_up(OFF_CONF + 4);                // B*K f32
constexpr size_t OFF_IDX  = align_up(OFF_GG + (size_t)BB * KK * 4);// B*K*N i32
constexpr size_t OFF_WL   = align_up(OFF_IDX + (size_t)BB * KK * NN * 4); // B*K*N f32

// ---------------- kernel: zero accumulators ----------------
__global__ void k_zero(float* z, int n) {
    int i = blockIdx.x * blockDim.x + threadIdx.x;
    if (i < n) z[i] = 0.0f;
}

// ---------------- kernel: geno gate contribution gg[b,k] ----------------
__global__ void k_gg(const float* __restrict__ geno, const float* __restrict__ Wgg,
                     const float* __restrict__ bgg, float* __restrict__ gg) {
    int t = threadIdx.x;
    if (t >= BB * KK) return;
    int b = t >> 3, k = t & 7;
    float acc = 0.f;
    for (int c = 0; c < CC; ++c) acc += geno[b * CC + c] * Wgg[k * CC + c];
    gg[t] = RATIO * (acc + bgg[k]);
}

// ---------------- kernel: W1 fp32 -> bf16 (vectorized) ----------------
__global__ void k_conv(const float* __restrict__ W1, __bf16* __restrict__ W1bf, int n8) {
    int i = blockIdx.x * blockDim.x + threadIdx.x;   // one v8 chunk per thread
    if (i >= n8) return;
    const float4* src = (const float4*)(W1 + (size_t)i * 8);
    float4 f0 = src[0], f1 = src[1];
    v8bf h;
    h[0] = (__bf16)f0.x; h[1] = (__bf16)f0.y; h[2] = (__bf16)f0.z; h[3] = (__bf16)f0.w;
    h[4] = (__bf16)f1.x; h[5] = (__bf16)f1.y; h[6] = (__bf16)f1.z; h[7] = (__bf16)f1.w;
    *(v8bf*)(W1bf + (size_t)i * 8) = h;
}

// ---------------- kernel: gating + routing lists ----------------
__global__ void k_gate(const float* __restrict__ tokens, const float* __restrict__ Wg,
                       const float* __restrict__ bg, const float* __restrict__ gg,
                       int* __restrict__ cnt, int* __restrict__ idxl, float* __restrict__ wl,
                       float* __restrict__ mass, int* __restrict__ usagecnt,
                       float* __restrict__ confsum) {
    __shared__ float WgL[KK * CC];
    __shared__ float cred[8];
    int tid = threadIdx.x;
    for (int i = tid; i < KK * CC; i += 256) WgL[i] = Wg[i];
    __syncthreads();

    int wv = tid >> 5, lane = tid & 31;
    int tok = blockIdx.x * 8 + wv;            // 0..B*N-1
    int b = tok >> 12, n = tok & (NN - 1);
    const float* x = tokens + (size_t)tok * CC;

    float acc[KK];
#pragma unroll
    for (int k = 0; k < KK; ++k) acc[k] = 0.f;
#pragma unroll
    for (int j = 0; j < 8; ++j) {
        int c = lane + j * 32;
        float xv = x[c];
#pragma unroll
        for (int k = 0; k < KK; ++k) acc[k] += xv * WgL[k * CC + c];
    }
#pragma unroll
    for (int k = 0; k < KK; ++k)
        for (int off = 16; off >= 1; off >>= 1) acc[k] += __shfl_xor(acc[k], off, 32);

    if (lane == 0) {
        float lg[KK];
#pragma unroll
        for (int k = 0; k < KK; ++k) lg[k] = (acc[k] + bg[k] + gg[b * KK + k]) * (1.0f / TEMPF);
        int i0 = 0; float v0 = lg[0];
#pragma unroll
        for (int k = 1; k < KK; ++k) if (lg[k] > v0) { v0 = lg[k]; i0 = k; }
        int i1 = -1; float v1 = -3.4e38f;
#pragma unroll
        for (int k = 0; k < KK; ++k) if (k != i0 && lg[k] > v1) { v1 = lg[k]; i1 = k; }
        // softmax over top-2, clamp to EPS, renormalize
        float e1 = __expf(v1 - v0);
        float w0 = 1.0f / (1.0f + e1);
        float w1 = e1 / (1.0f + e1);
        w0 = fmaxf(w0, EPSF); w1 = fmaxf(w1, EPSF);
        float inv = 1.0f / (w0 + w1);
        w0 *= inv; w1 *= inv;

        int bk0 = b * KK + i0, bk1 = b * KK + i1;
        int p0 = atomicAdd(&cnt[bk0], 1);
        idxl[(size_t)bk0 * NN + p0] = n; wl[(size_t)bk0 * NN + p0] = w0;
        int p1 = atomicAdd(&cnt[bk1], 1);
        idxl[(size_t)bk1 * NN + p1] = n; wl[(size_t)bk1 * NN + p1] = w1;
        atomicAdd(&mass[bk0], w0);
        atomicAdd(&mass[bk1], w1);
        atomicAdd(&usagecnt[i0], 1);
        atomicAdd(&usagecnt[i1], 1);
        cred[wv] = v0 - v1;
    }
    __syncthreads();
    if (tid == 0) {
        float s = 0.f;
#pragma unroll
        for (int i = 0; i < 8; ++i) s += cred[i];
        atomicAdd(confsum, s);
    }
}

// ---------------- kernel: expert layer-1 GEMM (WMMA bf16) + weighted reduce ----------------
// grid: (tiles=64, B*K); block 256 (8 waves). Tile = 64 routed tokens x 256 hidden.
__global__ void k_expert(const float* __restrict__ tokens, const __bf16* __restrict__ W1bf,
                         const float* __restrict__ b1g, const int* __restrict__ cnt,
                         const int* __restrict__ idxl, const float* __restrict__ wl,
                         float* __restrict__ sbuf) {
    int bk = blockIdx.y;
    int count = cnt[bk];
    int tilestart = blockIdx.x * 64;
    if (tilestart >= count) return;

    int b = bk >> 3, k = bk & 7;
    int tid = threadIdx.x;

    __shared__ __bf16 Xl[64 * 264];   // padded stride: conflict-free 16B LDS accesses
    __shared__ int    tks[64];
    __shared__ float  wt[64];

    if (tid < 64) {
        int rg = tilestart + tid;
        bool v = rg < count;
        tks[tid] = v ? idxl[(size_t)bk * NN + rg] : idxl[(size_t)bk * NN + tilestart];
        wt[tid]  = v ? wl[(size_t)bk * NN + rg] : 0.0f;
    }
    __syncthreads();

    // stage tile: 8 rows per pass; lane handles 8 contiguous floats -> one b128 LDS store
    {
        int rg = tid >> 5;             // row within group of 8
        int lc = (tid & 31) * 8;       // column base
#pragma unroll
        for (int r0 = 0; r0 < 64; r0 += 8) {
            int row = r0 + rg;
            const float4* src = (const float4*)(tokens + ((size_t)b * NN + tks[row]) * CC + lc);
            float4 f0 = src[0], f1 = src[1];
            v8bf h;
            h[0] = (__bf16)f0.x; h[1] = (__bf16)f0.y; h[2] = (__bf16)f0.z; h[3] = (__bf16)f0.w;
            h[4] = (__bf16)f1.x; h[5] = (__bf16)f1.y; h[6] = (__bf16)f1.z; h[7] = (__bf16)f1.w;
            *(v8bf*)&Xl[row * 264 + lc] = h;
        }
    }
    __syncthreads();

    int wv = tid >> 5, lane = tid & 31;
    int mbase = (wv >> 2) * 32;       // 0 or 32
    int nbase = (wv & 3) * 64;        // 0,64,128,192
    int hs = lane >> 4;

    const __bf16* Wk = W1bf + (size_t)k * CC * CC;

    v8f acc[2][4];
#pragma unroll
    for (int mi = 0; mi < 2; ++mi)
#pragma unroll
        for (int ni = 0; ni < 4; ++ni) acc[mi][ni] = (v8f)(0.0f);

    for (int kk = 0; kk < CC; kk += 32) {
        v16bf a[2];
#pragma unroll
        for (int mi = 0; mi < 2; ++mi) {
            int row = mbase + mi * 16 + (lane & 15);
            const __bf16* pr = &Xl[row * 264 + kk + hs * 16];
            union { v16bf v; v8bf h[2]; } u;
            u.h[0] = *(const v8bf*)pr;          // K = kk+hs*16 .. +7
            u.h[1] = *(const v8bf*)(pr + 8);    // K = kk+hs*16+8 .. +15
            a[mi] = u.v;
        }
        v16bf bb[4];
#pragma unroll
        for (int ni = 0; ni < 4; ++ni) {
            int j = nbase + ni * 16 + (lane & 15);
            bb[ni] = *(const v16bf*)&Wk[(size_t)j * CC + kk + hs * 16];
        }
        if (kk + 32 < CC)   // WGP-scope prefetch of next weight K-slice
            __builtin_prefetch(&Wk[(size_t)(nbase + (lane & 15)) * CC + kk + 32], 0, 3);
#pragma unroll
        for (int mi = 0; mi < 2; ++mi)
#pragma unroll
            for (int ni = 0; ni < 4; ++ni)
                acc[mi][ni] = __builtin_amdgcn_wmma_f32_16x16x32_bf16(
                    false, a[mi], false, bb[ni], (short)0, acc[mi][ni], false, false);
    }

    // h = relu(acc + b1); weighted sum over the 64 token rows; accumulate into sbuf[bk,:]
#pragma unroll
    for (int ni = 0; ni < 4; ++ni) {
        int j = nbase + ni * 16 + (lane & 15);
        float b1j = b1g[k * CC + j];
        float colsum = 0.f;
#pragma unroll
        for (int mi = 0; mi < 2; ++mi) {
            float p = 0.f;
#pragma unroll
            for (int g = 0; g < 8; ++g) {
                int M = mbase + mi * 16 + ((lane < 16) ? g : 8 + g);
                float h = fmaxf(acc[mi][ni][g] + b1j, 0.0f);
                p += wt[M] * h;
            }
            colsum += p;
        }
        colsum += __shfl_xor(colsum, 16, 32);
        if (lane < 16) atomicAdd(&sbuf[(size_t)bk * CC + j], colsum);
    }
}

// ---------------- kernel: centers = (s @ W2^T + mass*b2) / max(mass, eps) ----------------
__global__ void k_final(const float* __restrict__ sbuf, const float* __restrict__ mass,
                        const float* __restrict__ W2, const float* __restrict__ b2,
                        float* __restrict__ out) {
    int bk = blockIdx.x;
    int k = bk & 7, j = threadIdx.x;
    __shared__ float sl[CC];
    sl[j] = sbuf[(size_t)bk * CC + j];
    __syncthreads();
    float m = mass[bk];
    float md = fmaxf(m, EPSF);
    const float4* w2r = (const float4*)(W2 + ((size_t)k * CC + j) * CC);
    float acc = 0.f;
#pragma unroll 4
    for (int c4 = 0; c4 < CC / 4; ++c4) {
        float4 w = w2r[c4];
        acc += w.x * sl[c4 * 4] + w.y * sl[c4 * 4 + 1] + w.z * sl[c4 * 4 + 2] + w.w * sl[c4 * 4 + 3];
    }
    out[(size_t)bk * CC + j] = (acc + m * b2[k * CC + j]) / md;
}

// ---------------- kernel: lb_loss + routing confidence ----------------
__global__ void k_scalars(const int* __restrict__ usagecnt, const float* __restrict__ confsum,
                          float* __restrict__ out) {
    if (threadIdx.x != 0) return;
    float u[KK], s = 0.f;
#pragma unroll
    for (int k = 0; k < KK; ++k) { u[k] = (float)usagecnt[k] / (float)(BB * NN); s += u[k]; }
    float mean = s / KK, var = 0.f;
#pragma unroll
    for (int k = 0; k < KK; ++k) { float d = u[k] - mean; var += d * d; }
    var /= KK;
    float r = sqrtf(var) / (mean + EPSF);
    out[(size_t)BB * KK * CC + 0] = r * r;
    out[(size_t)BB * KK * CC + 1] = confsum[0] / (float)(BB * NN);
}

extern "C" void kernel_launch(void* const* d_in, const int* in_sizes, int n_in,
                              void* d_out, int out_size, void* d_ws, size_t ws_size,
                              hipStream_t stream) {
    const float* tokens = (const float*)d_in[0];
    const float* geno   = (const float*)d_in[1];
    const float* Wg     = (const float*)d_in[2];
    const float* bg     = (const float*)d_in[3];
    const float* Wgg    = (const float*)d_in[4];
    const float* bgg    = (const float*)d_in[5];
    const float* W1     = (const float*)d_in[6];
    const float* b1     = (const float*)d_in[7];
    const float* W2     = (const float*)d_in[8];
    const float* b2     = (const float*)d_in[9];
    float* out = (float*)d_out;

    char* ws = (char*)d_ws;
    __bf16* W1bf   = (__bf16*)(ws + OFF_W1BF);
    float*  sbuf   = (float*)(ws + OFF_SBUF);
    float*  mass   = (float*)(ws + OFF_MASS);
    int*    cnt    = (int*)(ws + OFF_CNT);
    int*    usagec = (int*)(ws + OFF_USE);
    float*  conf   = (float*)(ws + OFF_CONF);
    float*  gg     = (float*)(ws + OFF_GG);
    int*    idxl   = (int*)(ws + OFF_IDX);
    float*  wl     = (float*)(ws + OFF_WL);

    k_zero<<<(int)((ZERO_FLOATS + 255) / 256), 256, 0, stream>>>((float*)(ws + OFF_ZERO), (int)ZERO_FLOATS);
    k_gg<<<1, 128, 0, stream>>>(geno, Wgg, bgg, gg);
    k_conv<<<(KK * CC * CC / 8 + 255) / 256, 256, 0, stream>>>(W1, W1bf, KK * CC * CC / 8);
    k_gate<<<(BB * NN) / 8, 256, 0, stream>>>(tokens, Wg, bg, gg, cnt, idxl, wl, mass, usagec, conf);
    k_expert<<<dim3(NN / 64, BB * KK), 256, 0, stream>>>(tokens, W1bf, b1, cnt, idxl, wl, sbuf);
    k_final<<<BB * KK, CC, 0, stream>>>(sbuf, mass, W2, b2, out);
    k_scalars<<<1, 32, 0, stream>>>(usagec, conf, out);
}
